// VanillaRNNLayer_71382356460194
// MI455X (gfx1250) — compile-verified, matrix operand order checked by reading, not run on hardware
//
#include <hip/hip_runtime.h>
#include <hip/hip_bf16.h>
#include <math.h>

// ---------------------------------------------------------------------------
// Vanilla RNN layer for MI455X (gfx1250, wave32, WMMA).
//   Pass 0: convert x, Wx to bf16 once (x-bf16 = 64MB -> lives in 192MB L2).
//   Pass 1: x_proj = x @ Wx^T + bx + bh  (pure bf16-WMMA GEMM, fp32 accum).
//   Pass 2: persistent scan kernel: h <- tanh(xp_t + h @ Wh^T).
//           Wh slice in LDS (bf16), h state ping-pongs as bf16 in L2,
//           grid barrier per step, prefetch of next xp slab.
// ---------------------------------------------------------------------------

typedef __bf16 bf16;
typedef __attribute__((ext_vector_type(16))) __bf16 v16bf;
typedef __attribute__((ext_vector_type(8)))  __bf16 v8bf;
typedef __attribute__((ext_vector_type(4)))  __bf16 v4bf;
typedef __attribute__((ext_vector_type(8)))  float  v8f;
typedef __attribute__((ext_vector_type(4)))  float  v4f;

#define B_  32
#define T_  2048
#define I_  512
#define H_  512
#define NWG 16   // workgroups in the persistent scan kernel (H_/32 slices)

static constexpr size_t XP_ELEMS  = (size_t)T_ * B_ * H_;       // [T,B,H] f32
static constexpr size_t XP_BYTES  = XP_ELEMS * sizeof(float);   // 128 MB
static constexpr size_t XBF_BYTES = (size_t)B_ * T_ * I_ * 2;   // 64 MB
static constexpr size_t WXB_BYTES = (size_t)H_ * I_ * 2;        // 512 KB
static constexpr size_t HB_ELEMS  = (size_t)B_ * H_;            // one h buffer

// Load a 16x32 bf16 WMMA fragment from a row-major bf16 matrix.
// Layout (ISA 7.12.2, 16-bit A 16x32): lane l -> row (l&15);
//   elems 0..7  = K (l&16?8:0)+0..7,  elems 8..15 = K (l&16?8:0)+16..23.
// Also valid for B when B = W^T with W row-major [N,K]: column n of B is the
// contiguous row n of W.
__device__ __forceinline__ v16bf load_frag_bf16(const bf16* __restrict__ base,
                                                int ld, int row0, int k, int lane) {
  const int r   = row0 + (lane & 15);
  const int klo = k + ((lane & 16) ? 8 : 0);
  const bf16* p = base + (size_t)r * ld + klo;
  v8bf lo = *(const v8bf*)(p);        // global_load_b128 / ds_load_b128
  v8bf hi = *(const v8bf*)(p + 16);
  v16bf o;
#pragma unroll
  for (int i = 0; i < 8; ++i) { o[i] = lo[i]; o[i + 8] = hi[i]; }
  return o;
}

// ---------------------------------------------------------------------------
// Pass 0: vectorized f32 -> bf16 conversion (grid-stride, 4 elems/thread/iter)
// ---------------------------------------------------------------------------
__global__ __launch_bounds__(256) void f32_to_bf16_kernel(
    const float* __restrict__ src, bf16* __restrict__ dst, long n) {
  long i = ((long)blockIdx.x * blockDim.x + threadIdx.x) * 4;
  const long stride = (long)gridDim.x * blockDim.x * 4;
  for (; i < n; i += stride) {
    v4f f = *(const v4f*)(src + i);
    v4bf o;
#pragma unroll
    for (int j = 0; j < 4; ++j) o[j] = (bf16)f[j];
    *(v4bf*)(dst + i) = o;
  }
}

// ---------------------------------------------------------------------------
// Pass 1: x_proj[t][b][n] = sum_k xbf[b*T+t][k]*Wxbf[n][k] + bx[n] + bh[n]
// Block = 128 threads = 4 waves; wave tile 32(M) x 64(N) = 2x4 fragments,
// block tile 64(M) x 128(N). Bias folded into C init (all 8 C elements of a
// lane share N = lane&15). Inner loop: 12 b128 loads + 8 WMMAs, zero VALU.
// ---------------------------------------------------------------------------
__global__ __launch_bounds__(128) void rnn_xproj_gemm(
    const bf16* __restrict__ xbf,   // [B*T, I] bf16
    const bf16* __restrict__ wxb,   // [H, I]   bf16
    const float* __restrict__ bx,   // [H]
    const float* __restrict__ bh,   // [H]
    float* __restrict__ xp)         // [T, B, H]
{
  const int lane  = threadIdx.x & 31;
  const int wave  = threadIdx.x >> 5;      // 0..3
  const int warpM = wave & 1;
  const int warpN = wave >> 1;
  const int m0 = blockIdx.y * 64 + warpM * 32;
  const int n0 = blockIdx.x * 128 + warpN * 64;

  v8f c[2][4];
#pragma unroll
  for (int j = 0; j < 4; ++j) {
    const int n = n0 + j * 16 + (lane & 15);
    const float bias = bx[n] + bh[n];
#pragma unroll
    for (int i = 0; i < 2; ++i)
#pragma unroll
      for (int r = 0; r < 8; ++r) c[i][j][r] = bias;
  }

#pragma unroll 2
  for (int k = 0; k < I_; k += 32) {
    v16bf a0 = load_frag_bf16(xbf, I_, m0,      k, lane);
    v16bf a1 = load_frag_bf16(xbf, I_, m0 + 16, k, lane);
    v16bf b0 = load_frag_bf16(wxb, I_, n0,      k, lane);
    v16bf b1 = load_frag_bf16(wxb, I_, n0 + 16, k, lane);
    v16bf b2 = load_frag_bf16(wxb, I_, n0 + 32, k, lane);
    v16bf b3 = load_frag_bf16(wxb, I_, n0 + 48, k, lane);
    c[0][0] = __builtin_amdgcn_wmma_f32_16x16x32_bf16(false, a0, false, b0, (short)0, c[0][0], false, false);
    c[0][1] = __builtin_amdgcn_wmma_f32_16x16x32_bf16(false, a0, false, b1, (short)0, c[0][1], false, false);
    c[0][2] = __builtin_amdgcn_wmma_f32_16x16x32_bf16(false, a0, false, b2, (short)0, c[0][2], false, false);
    c[0][3] = __builtin_amdgcn_wmma_f32_16x16x32_bf16(false, a0, false, b3, (short)0, c[0][3], false, false);
    c[1][0] = __builtin_amdgcn_wmma_f32_16x16x32_bf16(false, a1, false, b0, (short)0, c[1][0], false, false);
    c[1][1] = __builtin_amdgcn_wmma_f32_16x16x32_bf16(false, a1, false, b1, (short)0, c[1][1], false, false);
    c[1][2] = __builtin_amdgcn_wmma_f32_16x16x32_bf16(false, a1, false, b2, (short)0, c[1][2], false, false);
    c[1][3] = __builtin_amdgcn_wmma_f32_16x16x32_bf16(false, a1, false, b3, (short)0, c[1][3], false, false);
  }

  // C element r of lane l -> M = r + (l&16?8:0), N = l&15.
  // Row m of flat x maps to (b = m/T, t = m%T); store time-major.
  const int rowhi = (lane & 16) ? 8 : 0;
#pragma unroll
  for (int i = 0; i < 2; ++i) {
#pragma unroll
    for (int r = 0; r < 8; ++r) {
      const int m = m0 + i * 16 + rowhi + r;
      const int b = m >> 11;        // / T_
      const int t = m & (T_ - 1);   // % T_
#pragma unroll
      for (int j = 0; j < 4; ++j) {
        const int n = n0 + j * 16 + (lane & 15);
        xp[((size_t)t * B_ + b) * H_ + n] = c[i][j][r];
      }
    }
  }
}

// ---------------------------------------------------------------------------
// Init: h buffer 0 <- bf16(h_0), grid-barrier counter <- 0 (per-call reset).
// ---------------------------------------------------------------------------
__global__ void rnn_init(const float* __restrict__ h0,
                         bf16* __restrict__ hbuf,
                         unsigned* __restrict__ bar) {
  const int i = blockIdx.x * blockDim.x + threadIdx.x;
  if (i < B_ * H_) hbuf[i] = (bf16)h0[i];
  if (i == 0) *bar = 0u;
}

// ---------------------------------------------------------------------------
// Pass 2: persistent scan. WG g owns columns [g*32, g*32+32). 128 threads =
// 4 waves; wave (mq,nq) computes the 16x16 fragment (rows mq*16.., cols
// nq*16.. of the slice). Wh slice cached in LDS as bf16 (32KB, loaded once).
// h state ping-pongs as bf16 through L2-resident ws buffers; monotonic
// atomic grid barrier per step; prefetch next xp slab during compute.
// ---------------------------------------------------------------------------
__global__ __launch_bounds__(128) void rnn_scan(
    const float* __restrict__ Wh,    // [H, H] f32
    const float* __restrict__ xp,    // [T, B, H] f32
    float* __restrict__ out,         // [B*T*H] outputs ++ [B*H] h_last
    bf16* __restrict__ hbuf,         // [2][B*H] bf16
    unsigned* __restrict__ bar)
{
  __shared__ __align__(16) bf16 ldsW[32 * H_];
  const int n0 = blockIdx.x * 32;

  for (int idx = threadIdx.x; idx < 32 * H_; idx += blockDim.x) {
    const int rr = idx >> 9;     // row within slice
    const int kk = idx & 511;
    ldsW[idx] = (bf16)Wh[(size_t)(n0 + rr) * H_ + kk];
  }
  __syncthreads();

  const int lane   = threadIdx.x & 31;
  const int wave   = threadIdx.x >> 5;   // 0..3
  const int mq     = wave & 1;           // batch-row half
  const int nq     = wave >> 1;          // column half of the slice
  const int rowhi  = (lane & 16) ? 8 : 0;
  const int ncol   = n0 + nq * 16 + (lane & 15);
  const bf16* __restrict__ wrow = &ldsW[(size_t)(nq * 16 + (lane & 15)) * H_];

  for (int t = 0; t < T_; ++t) {
    const bf16* __restrict__ hcur  = hbuf + (size_t)(t & 1) * HB_ELEMS;
    bf16* __restrict__       hnext = hbuf + (size_t)((t + 1) & 1) * HB_ELEMS;
    const float* __restrict__ xpt  = xp + (size_t)t * (B_ * H_);

    // Prefetch next step's x_proj slab (64KB, HBM-resident) while computing.
    if (t + 1 < T_) {
      const float* nxt = xpt + B_ * H_;
      __builtin_prefetch(nxt + (size_t)threadIdx.x * 8 +
                         (size_t)(blockIdx.x) * (B_ * H_ / NWG), 0, 1);
    }

    v8f c = {};
#pragma unroll 4
    for (int k = 0; k < H_; k += 32) {
      v16bf a = load_frag_bf16(hcur, H_, mq * 16, k, lane);  // global (L2)
      v16bf b = load_frag_bf16(wrow, 0, 0, k, lane);         // LDS (row fixed)
      c = __builtin_amdgcn_wmma_f32_16x16x32_bf16(false, a, false, b, (short)0, c, false, false);
    }

#pragma unroll
    for (int r = 0; r < 8; ++r) {
      const int b = mq * 16 + rowhi + r;
      const float v = tanhf(c[r] + xpt[(size_t)b * H_ + ncol]);
      out[((size_t)b * T_ + t) * H_ + ncol] = v;
      hnext[(size_t)b * H_ + ncol] = (bf16)v;
      if (t == T_ - 1)
        out[(size_t)B_ * T_ * H_ + (size_t)b * H_ + ncol] = v;
    }

    // ---- grid barrier (monotonic counter, no reset needed) ----
    __threadfence();
    __syncthreads();
    if (threadIdx.x == 0) {
      atomicAdd(bar, 1u);
      const unsigned target = (unsigned)(t + 1) * NWG;
      while (__hip_atomic_load(bar, __ATOMIC_RELAXED, __HIP_MEMORY_SCOPE_AGENT) < target)
        __builtin_amdgcn_s_sleep(2);
    }
    __syncthreads();
    __threadfence();
  }
}

// ---------------------------------------------------------------------------
// Workspace: xp (128MB) | x-bf16 (64MB) | Wx-bf16 (512KB) | h bufs | barrier.
// Requires ws_size >= ~192.6 MB.
// ---------------------------------------------------------------------------
extern "C" void kernel_launch(void* const* d_in, const int* in_sizes, int n_in,
                              void* d_out, int out_size, void* d_ws, size_t ws_size,
                              hipStream_t stream) {
  const float* x  = (const float*)d_in[0];
  const float* h0 = (const float*)d_in[1];
  const float* Wx = (const float*)d_in[2];
  const float* bx = (const float*)d_in[3];
  const float* Wh = (const float*)d_in[4];
  const float* bh = (const float*)d_in[5];
  float* out = (float*)d_out;

  char* ws = (char*)d_ws;
  float*    xp   = (float*)(ws);
  bf16*     xbf  = (bf16*)(ws + XP_BYTES);
  bf16*     wxb  = (bf16*)(ws + XP_BYTES + XBF_BYTES);
  bf16*     hbuf = (bf16*)(ws + XP_BYTES + XBF_BYTES + WXB_BYTES);
  unsigned* bar  = (unsigned*)(ws + XP_BYTES + XBF_BYTES + WXB_BYTES +
                               2 * HB_ELEMS * sizeof(bf16));

  rnn_init<<<(B_ * H_ + 255) / 256, 256, 0, stream>>>(h0, hbuf, bar);

  f32_to_bf16_kernel<<<4096, 256, 0, stream>>>(x,  xbf, (long)B_ * T_ * I_);
  f32_to_bf16_kernel<<<64,   256, 0, stream>>>(Wx, wxb, (long)H_ * I_);

  dim3 gA(H_ / 128, (B_ * T_) / 64);   // 4 x 1024 blocks
  rnn_xproj_gemm<<<gA, 128, 0, stream>>>(xbf, wxb, bx, bh, xp);

  rnn_scan<<<NWG, 128, 0, stream>>>(Wh, xp, out, hbuf, bar);
}
